// SparseDeltaModule_55250459296316
// MI455X (gfx1250) — matrix operand order repaired and checked by reading, not compiled, not run on hardware
//
#include <hip/hip_runtime.h>
#include <stdint.h>

// ---------------------------------------------------------------------------
// Sparse autoencoder forward for MI455X (gfx1250, wave32).
//   delta_hat [4096,2048], features [4096,16384] = SAE(X, W_enc, b_enc, W_dec, b_dec)
// Encoder GEMM (275 GFLOP) on v_wmma_f32_16x16x32_bf16. Both A (X) and B
// (W_enc) tiles are staged to LDS by the Tensor Data Mover (tensor_load_to_lds
// + s_wait_tensorcnt); K=64 per LDS stage -> 16 WMMAs per wave per barrier.
// Top-k is a 4-pass radix select; decode is a 64-way sparse gather from a
// pre-transposed, L2-resident W_dec.
// ---------------------------------------------------------------------------

#define NTOK    4096
#define DMODEL  2048
#define WIDTH_  16384
#define TOPK    64

typedef __bf16 bf16;
typedef __attribute__((ext_vector_type(16))) __bf16 v16bf;
typedef __attribute__((ext_vector_type(8)))  __bf16 v8bf;
typedef __attribute__((ext_vector_type(8)))  float  v8f;
typedef __attribute__((ext_vector_type(4)))  unsigned int v4u;
typedef __attribute__((ext_vector_type(4)))  int    v4i;
typedef __attribute__((ext_vector_type(8)))  int    v8i;

// -------------------------------- cast f32 -> bf16 --------------------------
__global__ __launch_bounds__(256) void cast_f32_to_bf16(const float* __restrict__ in,
                                                        bf16* __restrict__ out, int n) {
  int stride = gridDim.x * blockDim.x;
  for (int i = blockIdx.x * blockDim.x + threadIdx.x; i < n; i += stride)
    out[i] = (bf16)in[i];
}

// ------------------------- W_dec [D][W] -> W_decT [W][D] --------------------
__global__ __launch_bounds__(256) void transpose_wdec(const float* __restrict__ W,
                                                      float* __restrict__ WT) {
  __shared__ float tile[32][33];
  const int bx = blockIdx.x * 32;       // WIDTH index
  const int by = blockIdx.y * 32;       // DMODEL index
  const int tx = threadIdx.x & 31;
  const int ty = threadIdx.x >> 5;      // 8 row-groups
  for (int r = ty; r < 32; r += 8)
    tile[r][tx] = W[(size_t)(by + r) * WIDTH_ + bx + tx];
  __syncthreads();
  for (int r = ty; r < 32; r += 8)
    WT[(size_t)(bx + r) * DMODEL + by + tx] = tile[tx][r];
}

// ---------------------- TDM helper: 2D bf16 tile -> LDS ---------------------
#if __has_builtin(__builtin_amdgcn_tensor_load_to_lds)
#define HAVE_TDM 1
// Build the loop-invariant D# words once; only global_addr changes per stage.
struct TdmDesc {
  v4u g0;   // [0]=count, [1]=lds_addr, [2]=gaddr lo, [3]=gaddr hi | type
  v8i g1;   // dims / strides / tile shape
};

__device__ inline void tdm_init(TdmDesc& d, const void* lds_base,
                                int tensor_dim0, int tensor_dim1,
                                int tile_dim0, int tile_dim1, int stride0) {
  d.g0[0] = 1u;                                       // count=1, user mode
  d.g0[1] = (unsigned int)(uintptr_t)lds_base;        // LDS byte offset
  d.g0[2] = 0u;
  d.g0[3] = 0u;
  d.g1[0] = (int)(1u << 16);                          // data_size=1 -> 2 bytes
  d.g1[1] = (int)((tensor_dim0 & 0xFFFF) << 16);      // tensor_dim0 lo16 @48
  d.g1[2] = (int)((tensor_dim0 >> 16) |
                  ((tensor_dim1 & 0xFFFF) << 16));    // dim0 hi16 | dim1 lo16
  d.g1[3] = (int)((tensor_dim1 >> 16) |
                  ((tile_dim0 & 0xFFFF) << 16));      // dim1 hi16 | tile_dim0
  d.g1[4] = (int)(tile_dim1 & 0xFFFF);                // tile_dim1 (tile_dim2=0)
  d.g1[5] = stride0;                                  // tensor_dim0_stride lo32
  d.g1[6] = 0;
  d.g1[7] = 0;
}

__device__ inline void tdm_issue(TdmDesc& d, const bf16* gptr) {
  const unsigned long long ga = (unsigned long long)(uintptr_t)gptr;
  d.g0[2] = (unsigned int)(ga & 0xFFFFFFFFull);
  d.g0[3] = (unsigned int)((ga >> 32) & 0x1FFFFFFull) | (2u << 30); // type=2
  v4i z4 = {0, 0, 0, 0};
#if defined(__clang_major__) && (__clang_major__ >= 23)
  v8i z8 = {0, 0, 0, 0, 0, 0, 0, 0};
  __builtin_amdgcn_tensor_load_to_lds(d.g0, d.g1, z4, z4, z8, 0);
#else
  __builtin_amdgcn_tensor_load_to_lds(d.g0, d.g1, z4, z4, 0);
#endif
}
#else
#define HAVE_TDM 0
#endif

// ------------------------------- encoder GEMM -------------------------------
// dense[4096,16384] = relu(Xb[4096,2048] @ Wb[16384,2048]^T + b_enc)
// Block tile 128(M) x 128(N); 8 waves as 2x4; wave tile 64x32 = 4x2 WMMA tiles.
// K staged 64 at a time (two 16x16x32 sub-steps per stage).
#define KSTEP 64
__global__ __launch_bounds__(256) void encoder_gemm_wmma(
    const bf16*  __restrict__ Xb,     // [NTOK][DMODEL]
    const bf16*  __restrict__ Wb,     // [WIDTH_][DMODEL]
    const float* __restrict__ b_enc,  // [WIDTH_]
    float*       __restrict__ dense)  // [NTOK][WIDTH_]
{
  __shared__ __align__(128) bf16 Ash[128 * KSTEP];   // 16 KB  X tile    [m][k]
  __shared__ __align__(128) bf16 Bsh[128 * KSTEP];   // 16 KB  Wenc tile [n][k]

  const int tid  = threadIdx.x;
  const int lane = tid & 31;
  const int wid  = tid >> 5;
  const int block_n = blockIdx.x * 128;   // width tile
  const int block_m = blockIdx.y * 128;   // token tile
  const int wave_m  = (wid >> 2) * 64;    // 2 wave rows
  const int wave_n  = (wid & 3) * 32;     // 4 wave cols
  const int hi   = lane >> 4;             // 0: lanes 0-15, 1: lanes 16-31
  const int lrow = lane & 15;

  v8f acc[4][2] = {};

#if HAVE_TDM
  TdmDesc dA, dB;
  if (wid == 0) {
    tdm_init(dA, &Ash[0], DMODEL, NTOK,   KSTEP, 128, DMODEL);
    tdm_init(dB, &Bsh[0], DMODEL, WIDTH_, KSTEP, 128, DMODEL);
  }
#endif

  for (int k0 = 0; k0 < DMODEL; k0 += KSTEP) {
    __syncthreads();  // previous stage's LDS reads complete

#if HAVE_TDM
    if (wid == 0) {
      tdm_issue(dA, Xb + (size_t)block_m * DMODEL + k0);
      tdm_issue(dB, Wb + (size_t)block_n * DMODEL + k0);
#if __has_builtin(__builtin_amdgcn_s_wait_tensorcnt)
      __builtin_amdgcn_s_wait_tensorcnt(0);
#endif
    }
#else
    for (int c = tid; c < 1024; c += 256) {          // 1024 x 16B chunks / tile
      const int r = c >> 3, cc = c & 7;
      *(uint4*)(Ash + r * KSTEP + cc * 8) =
          *(const uint4*)(Xb + (size_t)(block_m + r) * DMODEL + k0 + cc * 8);
      *(uint4*)(Bsh + r * KSTEP + cc * 8) =
          *(const uint4*)(Wb + (size_t)(block_n + r) * DMODEL + k0 + cc * 8);
    }
#endif

    if (k0 + KSTEP < DMODEL)   // warm next W_enc tile (L2-resident weights)
      __builtin_prefetch(Wb + (size_t)(block_n + (tid & 127)) * DMODEL + k0 + KSTEP, 0, 1);

    __syncthreads();  // tiles visible to all 8 waves

    for (int s = 0; s < 2; ++s) {
      const int kk = s * 32;
      // A 16x32 bf16 frag: lanes 0-15 K{0..7,16..23}, lanes 16-31 K{8..15,24..31}
      v16bf afrag[4];
      for (int mi = 0; mi < 4; ++mi) {
        const bf16* ap = Ash + (wave_m + mi * 16 + lrow) * KSTEP + kk + hi * 8;
        v8bf c0 = *(const v8bf*)(ap);
        v8bf c1 = *(const v8bf*)(ap + 16);
        v16bf f;
        for (int e = 0; e < 8; ++e) { f[e] = c0[e]; f[e + 8] = c1[e]; }
        afrag[mi] = f;
      }
      // B 32x16 frag: column = lane%16, lanes 0-15 K{0..7,16..23}, lanes 16-31 K{8..15,24..31}
      v16bf bfrag[2];
      for (int ni = 0; ni < 2; ++ni) {
        const bf16* bp = Bsh + (wave_n + ni * 16 + lrow) * KSTEP + kk + hi * 8;
        v8bf c0 = *(const v8bf*)(bp);
        v8bf c1 = *(const v8bf*)(bp + 16);
        v16bf f;
        for (int e = 0; e < 8; ++e) { f[e] = c0[e]; f[e + 8] = c1[e]; }
        bfrag[ni] = f;
      }
      for (int mi = 0; mi < 4; ++mi)
        for (int ni = 0; ni < 2; ++ni)
          acc[mi][ni] = __builtin_amdgcn_wmma_f32_16x16x32_bf16(
              false, afrag[mi], false, bfrag[ni], (short)0, acc[mi][ni],
              false, false);
    }
  }

  // epilogue: bias + relu; C/D layout: VGPR r -> M = r + 8*hi, N = lane%16
  for (int mi = 0; mi < 4; ++mi)
    for (int ni = 0; ni < 2; ++ni) {
      const int col  = block_n + wave_n + ni * 16 + lrow;
      const float bb = b_enc[col];
      for (int r = 0; r < 8; ++r) {
        const int row = block_m + wave_m + mi * 16 + r + 8 * hi;
        float v = acc[mi][ni][r] + bb;
        dense[(size_t)row * WIDTH_ + col] = v > 0.f ? v : 0.f;
      }
    }
}

// ------------------------- per-row radix-select top-64 ----------------------
// Post-relu values are >= 0 so raw float bits are monotonic keys.
__global__ __launch_bounds__(256) void topk_kernel(float* __restrict__ dense,
                                                   int* __restrict__ tk_idx,
                                                   float* __restrict__ tk_val) {
  const int row = blockIdx.x;
  float* rp = dense + (size_t)row * WIDTH_;
  __shared__ unsigned int hist[256];
  __shared__ unsigned int s_prefix;
  __shared__ unsigned int s_remaining;
  __shared__ int s_nout, s_tie;
  const int tid = threadIdx.x;
  if (tid == 0) { s_prefix = 0; s_remaining = TOPK; s_nout = 0; s_tie = 0; }

  for (int shift = 24; shift >= 0; shift -= 8) {
    __syncthreads();
    hist[tid] = 0;
    __syncthreads();
    const unsigned int pfx = s_prefix;
    for (int i = tid; i < WIDTH_; i += 256) {
      const unsigned int k = __float_as_uint(rp[i]);
      const bool match = (shift == 24) || ((k >> (shift + 8)) == pfx);
      if (match) atomicAdd(&hist[(k >> shift) & 0xFF], 1u);
    }
    __syncthreads();
    if (tid == 0) {
      unsigned int rem = s_remaining, cum = 0;
      int bin = 0;
      for (int b = 255; b >= 0; --b) {
        const unsigned int c = hist[b];
        if (cum + c >= rem) { bin = b; rem -= cum; break; }
        cum += c;
      }
      s_prefix = (s_prefix << 8) | (unsigned int)bin;
      s_remaining = rem;
    }
  }
  __syncthreads();
  const unsigned int thresh  = s_prefix;
  const int          tiekeep = (int)s_remaining;

  for (int i = tid; i < WIDTH_; i += 256) {
    const float v = rp[i];
    const unsigned int k = __float_as_uint(v);
    bool keep = (k > thresh);
    if (!keep && k == thresh) keep = (atomicAdd(&s_tie, 1) < tiekeep);
    if (keep) {
      const int slot = atomicAdd(&s_nout, 1);
      tk_idx[row * TOPK + slot] = i;
      tk_val[row * TOPK + slot] = v;
    } else {
      rp[i] = 0.0f;
    }
  }
}

// ------------------------------ sparse decode -------------------------------
__global__ __launch_bounds__(256) void decode_kernel(
    const float* __restrict__ WT,      // [WIDTH_][DMODEL]  (transposed W_dec)
    const float* __restrict__ b_dec,   // [DMODEL]
    const int*   __restrict__ tk_idx,
    const float* __restrict__ tk_val,
    float*       __restrict__ delta)   // [NTOK][DMODEL]
{
  const int n   = blockIdx.x;
  const int tid = threadIdx.x;
  __shared__ int   s_idx[TOPK];
  __shared__ float s_val[TOPK];
  if (tid < TOPK) {
    s_idx[tid] = tk_idx[n * TOPK + tid];
    s_val[tid] = tk_val[n * TOPK + tid];
  }
  __syncthreads();

  float acc[8];
  for (int d = 0; d < 8; ++d) acc[d] = b_dec[tid + d * 256];
  for (int j = 0; j < TOPK; ++j) {
    const float* wrow = WT + (size_t)s_idx[j] * DMODEL;
    const float  v    = s_val[j];
    for (int d = 0; d < 8; ++d) acc[d] = fmaf(v, wrow[tid + d * 256], acc[d]);
  }
  for (int d = 0; d < 8; ++d)
    delta[(size_t)n * DMODEL + tid + d * 256] = acc[d];
}

// ------------------------------- launcher -----------------------------------
extern "C" void kernel_launch(void* const* d_in, const int* in_sizes, int n_in,
                              void* d_out, int out_size, void* d_ws, size_t ws_size,
                              hipStream_t stream) {
  (void)in_sizes; (void)n_in; (void)out_size; (void)ws_size;

  const float* X     = (const float*)d_in[0];  // [NTOK][DMODEL]
  const float* W_enc = (const float*)d_in[1];  // [WIDTH_][DMODEL]
  const float* b_enc = (const float*)d_in[2];  // [WIDTH_]
  const float* W_dec = (const float*)d_in[3];  // [DMODEL][WIDTH_]
  const float* b_dec = (const float*)d_in[4];  // [DMODEL]

  float* delta = (float*)d_out;                         // [NTOK][DMODEL]
  float* feats = (float*)d_out + (size_t)NTOK * DMODEL; // [NTOK][WIDTH_]

  // workspace layout (~220 MB)
  char* ws = (char*)d_ws;
  bf16*  Xb  = (bf16*)ws;  ws += (size_t)NTOK   * DMODEL * sizeof(bf16);
  bf16*  Wb  = (bf16*)ws;  ws += (size_t)WIDTH_ * DMODEL * sizeof(bf16);
  float* WT  = (float*)ws; ws += (size_t)WIDTH_ * DMODEL * sizeof(float);
  int*   tki = (int*)ws;   ws += (size_t)NTOK * TOPK * sizeof(int);
  float* tkv = (float*)ws;

  cast_f32_to_bf16<<<4096, 256, 0, stream>>>(X, Xb, NTOK * DMODEL);
  cast_f32_to_bf16<<<8192, 256, 0, stream>>>(W_enc, Wb, WIDTH_ * DMODEL);
  transpose_wdec<<<dim3(WIDTH_ / 32, DMODEL / 32), 256, 0, stream>>>(W_dec, WT);
  encoder_gemm_wmma<<<dim3(WIDTH_ / 128, NTOK / 128), 256, 0, stream>>>(Xb, Wb, b_enc, feats);
  topk_kernel<<<NTOK, 256, 0, stream>>>(feats, tki, tkv);
  decode_kernel<<<NTOK, 256, 0, stream>>>(WT, b_dec, tki, tkv, delta);
}
